// BlockSparseMoE_59777354826060
// MI455X (gfx1250) — matrix operand order repaired and checked by reading, not compiled, not run on hardware
//
#include <hip/hip_runtime.h>
#include <hip/hip_bf16.h>

typedef __attribute__((ext_vector_type(16))) __bf16 v16bf;
typedef __attribute__((ext_vector_type(8)))  float  v8f;

#define T_TOK 4096
#define DH    1024
#define DF    4096
#define NE    8

// ---- workspace layout (bytes) ----
#define OFF_CNT    0u
#define OFF_CURSOR 64u
#define OFF_OFFS   128u
#define OFF_TOPI   256u
#define OFF_TOPW   (OFF_TOPI + T_TOK*2*4u)
#define OFF_REC    (OFF_TOPW + T_TOK*2*4u)
#define OFF_WSL    (OFF_REC  + T_TOK*2*4u)
#define OFF_H      (OFF_WSL  + T_TOK*2*4u)                  // 131328, 16B aligned
#define OFF_OUTP   (OFF_H + (size_t)T_TOK*2*DF*2)           // + 64 MB bf16 h
// OUTP: (size_t)T_TOK*2*DH*4 = 32 MB fp32; total ~96.2 MB

__device__ inline unsigned pkbf(float a, float b) {
    union { __bf16 h[2]; unsigned u; } p;
    p.h[0] = (__bf16)a; p.h[1] = (__bf16)b;
    return p.u;
}

// ---------------- 0: init counters ----------------
__global__ void moe_init(int* cnt, int* cursor) {
    if (threadIdx.x < NE) { cnt[threadIdx.x] = 0; cursor[threadIdx.x] = 0; }
}

// ---------------- 1: router (one wave32 per token) ----------------
__global__ __launch_bounds__(256) void moe_router(
    const float* __restrict__ x, const float* __restrict__ gw,
    const float* __restrict__ gb, int* __restrict__ topi,
    float* __restrict__ topw, int* __restrict__ cnt)
{
    int lane  = threadIdx.x & 31;
    int token = blockIdx.x * 8 + (threadIdx.x >> 5);
    const float* xr = x + (size_t)token * DH;

    float acc[NE];
#pragma unroll
    for (int e = 0; e < NE; ++e) acc[e] = 0.f;

    for (int c = 0; c < DH / 32; ++c) {
        int d = c * 32 + lane;
        float xv = xr[d];
        const float4 g0 = *(const float4*)(gw + (size_t)d * NE);
        const float4 g1 = *(const float4*)(gw + (size_t)d * NE + 4);
        acc[0] += xv * g0.x; acc[1] += xv * g0.y; acc[2] += xv * g0.z; acc[3] += xv * g0.w;
        acc[4] += xv * g1.x; acc[5] += xv * g1.y; acc[6] += xv * g1.z; acc[7] += xv * g1.w;
    }
#pragma unroll
    for (int m = 16; m >= 1; m >>= 1)
#pragma unroll
        for (int e = 0; e < NE; ++e) acc[e] += __shfl_xor(acc[e], m, 32);

    if (lane == 0) {
        float l[NE];
        float mx = -1e30f;
#pragma unroll
        for (int e = 0; e < NE; ++e) { l[e] = acc[e] + gb[e]; mx = fmaxf(mx, l[e]); }
        float s = 0.f;
#pragma unroll
        for (int e = 0; e < NE; ++e) { l[e] = __expf(l[e] - mx); s += l[e]; }
        float inv = 1.f / s;
#pragma unroll
        for (int e = 0; e < NE; ++e) l[e] *= inv;

        int e0 = 0;
#pragma unroll
        for (int e = 1; e < NE; ++e) if (l[e] > l[e0]) e0 = e;
        int e1 = (e0 == 0) ? 1 : 0;
#pragma unroll
        for (int e = 0; e < NE; ++e) if (e != e0 && l[e] > l[e1]) e1 = e;

        float wsum = l[e0] + l[e1];
        topi[token * 2 + 0] = e0; topw[token * 2 + 0] = l[e0] / wsum;
        topi[token * 2 + 1] = e1; topw[token * 2 + 1] = l[e1] / wsum;
        atomicAdd(&cnt[e0], 1);
        atomicAdd(&cnt[e1], 1);
    }
}

// ---------------- 2: prefix offsets ----------------
__global__ void moe_offsets(const int* cnt, int* offs) {
    if (threadIdx.x == 0) {
        int s = 0;
        for (int e = 0; e < NE; ++e) { offs[e] = s; s += cnt[e]; }
    }
}

// ---------------- 3: scatter token slots ----------------
__global__ void moe_scatter(const int* __restrict__ topi, const float* __restrict__ topw,
                            const int* __restrict__ offs, int* __restrict__ cursor,
                            int* __restrict__ rec, float* __restrict__ wsl)
{
    int t = blockIdx.x * blockDim.x + threadIdx.x;
    if (t >= T_TOK) return;
#pragma unroll
    for (int j = 0; j < 2; ++j) {
        int e = topi[t * 2 + j];
        int pos = atomicAdd(&cursor[e], 1);
        int g = offs[e] + pos;
        rec[g] = t * 2 + j;
        wsl[g] = topw[t * 2 + j];
    }
}

// ---------------- 4: grouped GEMM1 + SwiGLU -> h (bf16) ----------------
// tile: M=32 (2 x 16 subtiles), N=128 (8 waves x 16), K-step 32
// grid: x = m-tile (fastest -> consecutive blocks share the B weight slice via L2)
__global__ __launch_bounds__(256) void moe_gemm1(
    const float* __restrict__ x, const float* __restrict__ w1,
    const float* __restrict__ w3, const int* __restrict__ rec,
    const int* __restrict__ cnt, const int* __restrict__ offs,
    __bf16* __restrict__ hbuf)
{
    const int e  = blockIdx.z;
    const int m0 = blockIdx.x * 32;
    const int n0 = blockIdx.y * 128;
    const int cnte = cnt[e];
    if (m0 >= cnte) return;
    const int base = offs[e];

    __shared__ alignas(16) __bf16 sA[32][32];
    __shared__ alignas(16) __bf16 sB1[32][128];
    __shared__ alignas(16) __bf16 sB3[32][128];
    __shared__ int sTok[32];

    const int tid  = threadIdx.x;
    const int lane = tid & 31;
    const int wave = tid >> 5;
    const int nw   = wave * 16;          // wave's N sub-tile inside sB

    if (tid < 32) {
        int slot = m0 + tid;
        sTok[tid] = (slot < cnte) ? (rec[base + slot] >> 1) : -1;
    }
    __syncthreads();

    const float* w1p = w1 + (size_t)e * DH * DF;
    const float* w3p = w3 + (size_t)e * DH * DF;

    v8f acc1a = {}, acc1b = {}, acc3a = {}, acc3b = {};

    for (int k0 = 0; k0 < DH; k0 += 32) {
        // stage A: 32x32 fp32 -> bf16 (one float4 per thread)
        {
            int r = tid >> 3, c = (tid & 7) * 4;
            int t = sTok[r];
            float4 v = make_float4(0.f, 0.f, 0.f, 0.f);
            if (t >= 0) v = *(const float4*)(x + (size_t)t * DH + k0 + c);
            *(unsigned*)&sA[r][c]     = pkbf(v.x, v.y);
            *(unsigned*)&sA[r][c + 2] = pkbf(v.z, v.w);
        }
        // stage B1/B3: 32x128 fp32 each -> bf16 (4 float4 per thread per matrix)
#pragma unroll
        for (int i = 0; i < 4; ++i) {
            int idx = tid + i * 256;          // 0..1023
            int r = idx >> 5;                 // 0..31
            int c = (idx & 31) * 4;           // 0..124
            const float* p1 = w1p + (size_t)(k0 + r) * DF + n0 + c;
            const float* p3 = w3p + (size_t)(k0 + r) * DF + n0 + c;
            float4 a = *(const float4*)p1;
            float4 b = *(const float4*)p3;
            *(unsigned*)&sB1[r][c]     = pkbf(a.x, a.y);
            *(unsigned*)&sB1[r][c + 2] = pkbf(a.z, a.w);
            *(unsigned*)&sB3[r][c]     = pkbf(b.x, b.y);
            *(unsigned*)&sB3[r][c + 2] = pkbf(b.z, b.w);
            if (k0 + 32 < DH) {               // prefetch next K-step of weights
                __builtin_prefetch(p1 + (size_t)32 * DF, 0, 0);
                __builtin_prefetch(p3 + (size_t)32 * DF, 0, 0);
            }
        }
        __syncthreads();

        // fragments per documented wave32 layouts
        union { v16bf v; uint4 q[2]; } a0, a1, b1f, b3f;
        int ar = lane & 15, kb = (lane >> 4) * 8;
        a0.q[0]  = *(const uint4*)&sA[ar][kb];
        a0.q[1]  = *(const uint4*)&sA[ar][kb + 16];
        a1.q[0]  = *(const uint4*)&sA[16 + ar][kb];
        a1.q[1]  = *(const uint4*)&sA[16 + ar][kb + 16];
#pragma unroll
        for (int s = 0; s < 16; ++s) {
            int kk = kb + (s & 7) + (s >> 3) * 16;
            b1f.v[s] = sB1[kk][nw + ar];
            b3f.v[s] = sB3[kk][nw + ar];
        }

        acc1a = __builtin_amdgcn_wmma_f32_16x16x32_bf16(false, a0.v, false, b1f.v,
                                                        (short)0, acc1a, false, false);
        acc1b = __builtin_amdgcn_wmma_f32_16x16x32_bf16(false, a1.v, false, b1f.v,
                                                        (short)0, acc1b, false, false);
        acc3a = __builtin_amdgcn_wmma_f32_16x16x32_bf16(false, a0.v, false, b3f.v,
                                                        (short)0, acc3a, false, false);
        acc3b = __builtin_amdgcn_wmma_f32_16x16x32_bf16(false, a1.v, false, b3f.v,
                                                        (short)0, acc3b, false, false);
        __syncthreads();
    }

    // epilogue: h = silu(g) * u, store bf16
    const int mb = (lane >> 4) * 8;
    const int nn = n0 + nw + (lane & 15);
#pragma unroll
    for (int v = 0; v < 8; ++v) {
        int slot0 = m0 + mb + v;
        if (slot0 < cnte) {
            float g = acc1a[v];
            float sg = g / (1.f + __expf(-g));
            hbuf[(size_t)(base + slot0) * DF + nn] = (__bf16)(sg * acc3a[v]);
        }
        int slot1 = m0 + 16 + mb + v;
        if (slot1 < cnte) {
            float g = acc1b[v];
            float sg = g / (1.f + __expf(-g));
            hbuf[(size_t)(base + slot1) * DF + nn] = (__bf16)(sg * acc3b[v]);
        }
    }
}

// ---------------- 5: grouped GEMM2: h @ w2, scale by routing weight ----------------
// tile: M=32, N=128, K-step 32; A tile staged with async global->LDS copies
__global__ __launch_bounds__(256) void moe_gemm2(
    const __bf16* __restrict__ hbuf, const float* __restrict__ w2,
    const int* __restrict__ rec, const int* __restrict__ cnt,
    const int* __restrict__ offs, const float* __restrict__ wsl,
    float* __restrict__ outpair)
{
    const int e  = blockIdx.z;
    const int m0 = blockIdx.x * 32;
    const int n0 = blockIdx.y * 128;
    const int cnte = cnt[e];
    if (m0 >= cnte) return;
    const int base = offs[e];

    __shared__ alignas(16) __bf16 sA[32][32];
    __shared__ alignas(16) __bf16 sB[32][128];

    const int tid  = threadIdx.x;
    const int lane = tid & 31;
    const int wave = tid >> 5;
    const int nw   = wave * 16;

    const float* w2p = w2 + (size_t)e * DF * DH;
    v8f acca = {}, accb = {};

    for (int k0 = 0; k0 < DF; k0 += 32) {
        // stage A (already bf16): 32x32 halves = 128 x 16B, async copy to LDS
        if (tid < 128) {
            int r = tid >> 2, c = (tid & 3) * 8;
            int slot = m0 + r;
            __bf16* lp = &sA[r][c];
            if (slot < cnte) {
                const __bf16* gp = hbuf + (size_t)(base + slot) * DF + k0 + c;
                unsigned lds_off = (unsigned)(uintptr_t)lp;
                asm volatile("global_load_async_to_lds_b128 %0, %1, off"
                             :: "v"(lds_off), "v"(gp) : "memory");
            } else {
                *(uint4*)lp = make_uint4(0u, 0u, 0u, 0u);
            }
        }
        // stage B: 32x128 fp32 -> bf16
#pragma unroll
        for (int i = 0; i < 4; ++i) {
            int idx = tid + i * 256;
            int r = idx >> 5;
            int c = (idx & 31) * 4;
            const float* p = w2p + (size_t)(k0 + r) * DH + n0 + c;
            float4 a = *(const float4*)p;
            *(unsigned*)&sB[r][c]     = pkbf(a.x, a.y);
            *(unsigned*)&sB[r][c + 2] = pkbf(a.z, a.w);
            if (k0 + 32 < DF) __builtin_prefetch(p + (size_t)32 * DH, 0, 0);
        }
        asm volatile("s_wait_asynccnt 0x0" ::: "memory");
        __syncthreads();

        union { v16bf v; uint4 q[2]; } a0, a1, bf;
        int ar = lane & 15, kb = (lane >> 4) * 8;
        a0.q[0] = *(const uint4*)&sA[ar][kb];
        a0.q[1] = *(const uint4*)&sA[ar][kb + 16];
        a1.q[0] = *(const uint4*)&sA[16 + ar][kb];
        a1.q[1] = *(const uint4*)&sA[16 + ar][kb + 16];
#pragma unroll
        for (int s = 0; s < 16; ++s)
            bf.v[s] = sB[kb + (s & 7) + (s >> 3) * 16][nw + ar];

        acca = __builtin_amdgcn_wmma_f32_16x16x32_bf16(false, a0.v, false, bf.v,
                                                       (short)0, acca, false, false);
        accb = __builtin_amdgcn_wmma_f32_16x16x32_bf16(false, a1.v, false, bf.v,
                                                       (short)0, accb, false, false);
        __syncthreads();
    }

    const int mb = (lane >> 4) * 8;
    const int nn = n0 + nw + (lane & 15);
#pragma unroll
    for (int v = 0; v < 8; ++v) {
        int slot0 = m0 + mb + v;
        if (slot0 < cnte) {
            int r   = rec[base + slot0];     // t*2 + j
            float w = wsl[base + slot0];
            outpair[(size_t)r * DH + nn] = w * acca[v];
        }
        int slot1 = m0 + 16 + mb + v;
        if (slot1 < cnte) {
            int r   = rec[base + slot1];
            float w = wsl[base + slot1];
            outpair[(size_t)r * DH + nn] = w * accb[v];
        }
    }
}

// ---------------- 6: combine the two expert contributions ----------------
__global__ void moe_combine(const float* __restrict__ outpair, float* __restrict__ out)
{
    int i = blockIdx.x * blockDim.x + threadIdx.x;   // i = t*DH + d
    if (i >= T_TOK * DH) return;
    int t = i >> 10, d = i & (DH - 1);
    out[i] = outpair[(size_t)(2 * t) * DH + d] + outpair[(size_t)(2 * t + 1) * DH + d];
}

// ---------------- host launch ----------------
extern "C" void kernel_launch(void* const* d_in, const int* in_sizes, int n_in,
                              void* d_out, int out_size, void* d_ws, size_t ws_size,
                              hipStream_t stream)
{
    const float* x  = (const float*)d_in[0];
    const float* gw = (const float*)d_in[1];
    const float* gb = (const float*)d_in[2];
    const float* w1 = (const float*)d_in[3];
    const float* w3 = (const float*)d_in[4];
    const float* w2 = (const float*)d_in[5];
    float* out = (float*)d_out;

    char* ws = (char*)d_ws;
    int*    cnt    = (int*)(ws + OFF_CNT);
    int*    cursor = (int*)(ws + OFF_CURSOR);
    int*    offs   = (int*)(ws + OFF_OFFS);
    int*    topi   = (int*)(ws + OFF_TOPI);
    float*  topw   = (float*)(ws + OFF_TOPW);
    int*    rec    = (int*)(ws + OFF_REC);
    float*  wsl    = (float*)(ws + OFF_WSL);
    __bf16* hbuf   = (__bf16*)(ws + OFF_H);
    float*  outp   = (float*)(ws + OFF_OUTP);

    moe_init<<<1, 32, 0, stream>>>(cnt, cursor);
    moe_router<<<T_TOK / 8, 256, 0, stream>>>(x, gw, gb, topi, topw, cnt);
    moe_offsets<<<1, 32, 0, stream>>>(cnt, offs);
    moe_scatter<<<(T_TOK + 255) / 256, 256, 0, stream>>>(topi, topw, offs, cursor, rec, wsl);

    // m-tile fastest so consecutive blocks reuse the same weight slice out of L2
    dim3 g1(T_TOK / 32, DF / 128, NE);       // (128, 32, 8); inactive tiles early-exit
    moe_gemm1<<<g1, 256, 0, stream>>>(x, w1, w3, rec, cnt, offs, hbuf);

    dim3 g2(T_TOK / 32, DH / 128, NE);       // (128, 8, 8)
    moe_gemm2<<<g2, 256, 0, stream>>>(hbuf, w2, rec, cnt, offs, wsl, outp);

    moe_combine<<<(T_TOK * DH + 255) / 256, 256, 0, stream>>>(outp, out);
}
